// MambaPasswordModel_20804821582031
// MI455X (gfx1250) — matrix operand order repaired and checked by reading, not compiled, not run on hardware
//
#include <hip/hip_runtime.h>
#include <hip/hip_bf16.h>

// ---- model dimensions (match reference) ----
constexpr int B_   = 4;
constexpr int S_   = 1024;
constexpr int D_   = 512;
constexpr int L_   = 4;
constexpr int DS_  = 16;
constexpr int DC_  = 4;
constexpr int V_   = 128;
constexpr int LAT_ = 64;
constexpr int DI_  = 2 * D_;        // 1024
constexpr int ROWS = B_ * S_;       // 4096 tokens
constexpr int XZW  = 2 * DI_;       // 2048
constexpr int XDW  = 2 * DS_ + 1;   // 33

typedef __attribute__((ext_vector_type(16))) __bf16 bf16x16;
typedef __attribute__((ext_vector_type(8)))  float  f32x8;

union FragBF { uint4 u[2]; bf16x16 v; };

__device__ __forceinline__ unsigned short f2bf(float f) {
  unsigned u = __float_as_uint(f);
  u += 0x7FFFu + ((u >> 16) & 1u);          // round-to-nearest-even
  return (unsigned short)(u >> 16);
}

// ---------------- fp32 -> bf16 weight conversion ----------------
__global__ void __launch_bounds__(256) f32_to_bf16_kernel(
    const float* __restrict__ in, unsigned short* __restrict__ out, int n) {
  int i = blockIdx.x * 256 + threadIdx.x;
  if (i < n) out[i] = f2bf(in[i]);
}

// ---------------- conditioning vector: cond = latent @ cond_w.T + cond_b ----
__global__ void __launch_bounds__(256) cond_kernel(
    const float* __restrict__ latent, const float* __restrict__ cw,
    const float* __restrict__ cb, float* __restrict__ cond) {
  int i = blockIdx.x * 256 + threadIdx.x;          // over B*D
  if (i >= B_ * D_) return;
  int d = i % D_, b = i / D_;
  const float* lr = latent + b * LAT_;
  const float* wr = cw + (size_t)d * LAT_;
  float acc = cb[d];
#pragma unroll 4
  for (int k = 0; k < LAT_; ++k) acc = fmaf(lr[k], wr[k], acc);
  cond[i] = acc;
}

// ---------------- embedding: x = tok_emb[ids] + pos_emb + cond ----------------
__global__ void __launch_bounds__(256) embed_kernel(
    const int* __restrict__ ids, const float* __restrict__ tok,
    const float* __restrict__ pos, const float* __restrict__ cond,
    float* __restrict__ x) {
  int i = blockIdx.x * 256 + threadIdx.x;          // over B*S*D (exact)
  int d = i % D_;
  int s = (i / D_) % S_;
  int b = i / (D_ * S_);
  int id = ids[b * S_ + s];
  x[i] = tok[(size_t)id * D_ + d] + pos[(size_t)s * D_ + d] + cond[b * D_ + d];
}

// ---------------- layernorm (one wave per row) -> bf16 ----------------
__global__ void __launch_bounds__(256) layernorm_bf16_kernel(
    const float* __restrict__ x, const float* __restrict__ w,
    const float* __restrict__ b, unsigned short* __restrict__ out) {
  int row  = blockIdx.x * 8 + (threadIdx.x >> 5);
  int lane = threadIdx.x & 31;
  const float* xr = x + (size_t)row * D_;
  float s = 0.f, ss = 0.f;
#pragma unroll
  for (int i = lane; i < D_; i += 32) { float v = xr[i]; s += v; ss = fmaf(v, v, ss); }
#pragma unroll
  for (int off = 16; off > 0; off >>= 1) {
    s  += __shfl_xor(s,  off, 32);
    ss += __shfl_xor(ss, off, 32);
  }
  const float rD = 1.f / (float)D_;
  float mu  = s * rD;
  float inv = rsqrtf(ss * rD - mu * mu + 1e-5f);
#pragma unroll
  for (int i = lane; i < D_; i += 32)
    out[(size_t)row * D_ + i] = f2bf((xr[i] - mu) * inv * w[i] + b[i]);
}

// ---------------- WMMA GEMM: C[M,N] = A[M,K](bf16) @ W[N,K](bf16)^T (+res) ----
// block = 256 threads = 8 waves along M; all waves share one 64-col N strip.
// The 64x64 bf16 B slab (8KB) is staged in LDS with CDNA5 async copies
// (global_load_async_to_lds_b128, ASYNCcnt) and double-buffered so copies for
// K-step k+64 overlap the 8 WMMAs of step k. Each thread owns two 16B slices
// per buffer (wait asynccnt<=2 drains the older buffer; async loads complete
// in order). B fragments use distinct VGPRs so all 8 ds_load_b128 issue as a
// clause before the WMMAs. A fragments are per-wave private, loaded straight
// to VGPRs before the barrier so their latency hides under the sync.
__global__ void __launch_bounds__(256) gemm_bf16_wmma(
    const unsigned short* __restrict__ A, const unsigned short* __restrict__ W,
    const float* __restrict__ residual, float* __restrict__ C,
    int M, int N, int K) {
  __shared__ __align__(16) unsigned short tileB[2][64 * 64];   // 2 x 8KB

  const int tid   = threadIdx.x;
  const int lane  = tid & 31;
  const int wave  = tid >> 5;
  const int half  = lane >> 4;                  // 0: lanes 0-15, 1: lanes 16-31
  const int l15   = lane & 15;
  const int tileM = blockIdx.x * 128 + wave * 16;
  const int tileN = blockIdx.y * 64;

  // two 16B slices per thread of the 64(row) x 64(K-half) B slab
  const int r0 = tid >> 3;                      // rows 0..31
  const int h0 = (tid & 7) * 8;                 // halfword offset in row
  const int r1 = (tid + 256) >> 3;              // rows 32..63
  const int h1 = h0;
  const unsigned short* gsrc0 = W + (size_t)(tileN + r0) * K + h0;
  const unsigned short* gsrc1 = W + (size_t)(tileN + r1) * K + h1;
  const unsigned ldsDst0[2] = {
      (unsigned)(size_t)&tileB[0][r0 * 64 + h0],
      (unsigned)(size_t)&tileB[1][r0 * 64 + h0]};
  const unsigned ldsDst1[2] = {
      (unsigned)(size_t)&tileB[0][r1 * 64 + h1],
      (unsigned)(size_t)&tileB[1][r1 * 64 + h1]};

  const unsigned short* Arow = A + (size_t)(tileM + l15) * K;
  const int aoff = half * 8;    // A lanes 16-31 carry K octets {8..15, 24..31}
  const int boff = half * 8;    // B lanes 16-31 carry K octets {8..15, 24..31}

  f32x8 acc[4] = {f32x8{}, f32x8{}, f32x8{}, f32x8{}};

  auto issue_copy = [&](int buf, int k0) {
    const unsigned short* g0 = gsrc0 + k0;
    const unsigned short* g1 = gsrc1 + k0;
    asm volatile("global_load_async_to_lds_b128 %0, %1, off"
                 :: "v"(ldsDst0[buf]), "v"(g0) : "memory");
    asm volatile("global_load_async_to_lds_b128 %0, %1, off"
                 :: "v"(ldsDst1[buf]), "v"(g1) : "memory");
  };

  issue_copy(0, 0);   // prologue

  for (int k0 = 0; k0 < K; k0 += 64) {
    const int  cur  = (k0 >> 6) & 1;
    const bool more = (k0 + 64) < K;

    if (more) issue_copy(cur ^ 1, k0 + 64);     // prefetch next slab

    FragBF a0, a1;    // per-wave A fragments straight from global
    a0.u[0] = *(const uint4*)(Arow + k0 + aoff);
    a0.u[1] = *(const uint4*)(Arow + k0 + aoff + 16);
    a1.u[0] = *(const uint4*)(Arow + k0 + 32 + aoff);
    a1.u[1] = *(const uint4*)(Arow + k0 + 32 + aoff + 16);

    // drain only the older pair of async copies, then fence all waves
    if (more) asm volatile("s_wait_asynccnt 2" ::: "memory");
    else      asm volatile("s_wait_asynccnt 0" ::: "memory");
    __syncthreads();

#pragma unroll
    for (int kk = 0; kk < 2; ++kk) {
      FragBF bfr[4];  // distinct VGPRs -> all 8 ds_loads clause together
#pragma unroll
      for (int t = 0; t < 4; ++t) {
        const uint4* bp =
            (const uint4*)&tileB[cur][(16 * t + l15) * 64 + kk * 32 + boff];
        bfr[t].u[0] = bp[0];
        bfr[t].u[1] = bp[2];
      }
#pragma unroll
      for (int t = 0; t < 4; ++t) {
        acc[t] = __builtin_amdgcn_wmma_f32_16x16x32_bf16(
            false, kk ? a1.v : a0.v, false, bfr[t].v, (short)0, acc[t],
            false, false);
      }
    }
    __syncthreads();   // all waves done reading before this buffer is rewritten
  }

  const int rowBase = tileM + half * 8;         // C/D layout: VGPR r -> row r (+8)
#pragma unroll
  for (int t = 0; t < 4; ++t) {
    const int col = tileN + t * 16 + l15;
#pragma unroll
    for (int r = 0; r < 8; ++r) {
      size_t idx = (size_t)(rowBase + r) * N + col;
      float v = acc[t][r];
      if (residual) v += residual[idx];
      C[idx] = v;
    }
  }
}

// ---------------- depthwise causal conv (DC=4) + bias + SiLU ----------------
__global__ void __launch_bounds__(256) conv_silu_kernel(
    const float* __restrict__ xz, const float* __restrict__ cw,
    const float* __restrict__ cb, float* __restrict__ xc) {
  int i = blockIdx.x * 256 + threadIdx.x;       // over B*S*DI (exact)
  int c = i % DI_;
  int s = (i / DI_) % S_;
  int b = i / (DI_ * S_);
  const float* w = cw + c * DC_;
  float acc = cb[c];
#pragma unroll
  for (int j = 0; j < DC_; ++j) {
    int sp = s + j - (DC_ - 1);
    if (sp >= 0) acc = fmaf(xz[(size_t)(b * S_ + sp) * XZW + c], w[j], acc);
  }
  xc[i] = acc * (1.f / (1.f + __expf(-acc)));   // silu
}

// ---------------- small projection: xd[r, 0..32] = xc[r] @ xp_w^T ----------------
__global__ void __launch_bounds__(256) xproj_kernel(
    const float* __restrict__ xc, const float* __restrict__ xpw,
    float* __restrict__ xd) {
  int i = blockIdx.x * 256 + threadIdx.x;       // over ROWS*33
  if (i >= ROWS * XDW) return;
  int o  = i % XDW;
  int rs = i / XDW;
  const float4* xr = (const float4*)(xc  + (size_t)rs * DI_);
  const float4* wr = (const float4*)(xpw + (size_t)o  * DI_);
  float acc = 0.f;
#pragma unroll 4
  for (int k = 0; k < DI_ / 4; ++k) {
    float4 a = xr[k], w = wr[k];
    acc = fmaf(a.x, w.x, fmaf(a.y, w.y, fmaf(a.z, w.z, fmaf(a.w, w.w, acc))));
  }
  xd[i] = acc;
}

// ---------------- sequential selective scan, fused softplus / SiLU(z) gate ----
// one thread per (b, channel); h[16] lives in registers; emits bf16 y for GEMM.
// delta/B/C rows are uniform within a wave -> broadcast loads out of L2.
__global__ void __launch_bounds__(256) scan_kernel(
    const float* __restrict__ xc, const float* __restrict__ xd,
    const float* __restrict__ xz, const float* __restrict__ A_log,
    unsigned short* __restrict__ ybf) {
  int t = blockIdx.x * 256 + threadIdx.x;       // over B*DI = 4096
  int c = t % DI_;
  int b = t / DI_;
  float A[DS_], h[DS_];
#pragma unroll
  for (int n = 0; n < DS_; ++n) { A[n] = -__expf(A_log[c * DS_ + n]); h[n] = 0.f; }
  for (int s = 0; s < S_; ++s) {
    size_t rs = (size_t)(b * S_ + s);
    const float* row = xd + rs * XDW;
    float dr    = row[0];
    float delta = (dr > 20.f) ? dr : log1pf(__expf(dr));   // softplus
    float du    = delta * xc[rs * DI_ + c];
    float y = 0.f;
#pragma unroll
    for (int n = 0; n < DS_; ++n) {
      float dA = __expf(delta * A[n]);
      h[n] = fmaf(dA, h[n], du * row[1 + n]);              // h = dA*h + dt*B*u
      y    = fmaf(h[n], row[1 + DS_ + n], y);              // y += h*C
    }
    float zv = xz[rs * XZW + DI_ + c];
    float g  = zv * (1.f / (1.f + __expf(-zv)));           // silu(z)
    ybf[rs * DI_ + c] = f2bf(y * g);
  }
}

// ---------------- host-side orchestration ----------------
extern "C" void kernel_launch(void* const* d_in, const int* in_sizes, int n_in,
                              void* d_out, int out_size, void* d_ws, size_t ws_size,
                              hipStream_t stream) {
  (void)in_sizes; (void)n_in; (void)out_size; (void)ws_size;
  const int*   ids     = (const int*)  d_in[0];
  const float* latent  = (const float*)d_in[1];
  const float* tok_emb = (const float*)d_in[2];
  const float* pos_emb = (const float*)d_in[3];
  const float* cond_w  = (const float*)d_in[4];
  const float* cond_b  = (const float*)d_in[5];
  const float* ln_w    = (const float*)d_in[6];
  const float* ln_b    = (const float*)d_in[7];
  const float* in_w    = (const float*)d_in[8];
  const float* conv_w  = (const float*)d_in[9];
  const float* conv_b  = (const float*)d_in[10];
  const float* A_log   = (const float*)d_in[11];
  const float* xp_w    = (const float*)d_in[12];
  const float* out_w   = (const float*)d_in[13];
  const float* fln_w   = (const float*)d_in[14];
  const float* fln_b   = (const float*)d_in[15];
  const float* lm_w    = (const float*)d_in[16];
  float* out = (float*)d_out;

  char* ws = (char*)d_ws;
  size_t off = 0;
  auto take = [&](size_t bytes) -> void* {
    void* p = ws + off;
    off = (off + bytes + 255) & ~(size_t)255;
    return p;
  };
  unsigned short* w_in_bf  = (unsigned short*)take(sizeof(short) * (size_t)L_ * 2 * DI_ * D_);
  unsigned short* w_out_bf = (unsigned short*)take(sizeof(short) * (size_t)L_ * D_ * DI_);
  unsigned short* w_lm_bf  = (unsigned short*)take(sizeof(short) * (size_t)V_ * D_);
  float*          cond     = (float*)         take(sizeof(float) * B_ * D_);
  float*          x        = (float*)         take(sizeof(float) * (size_t)ROWS * D_);
  unsigned short* ln_bf    = (unsigned short*)take(sizeof(short) * (size_t)ROWS * D_);
  float*          xz       = (float*)         take(sizeof(float) * (size_t)ROWS * XZW);
  float*          xc       = (float*)         take(sizeof(float) * (size_t)ROWS * DI_);
  float*          xd       = (float*)         take(sizeof(float) * (size_t)ROWS * XDW);
  unsigned short* y_bf     = (unsigned short*)take(sizeof(short) * (size_t)ROWS * DI_);

  // weights -> bf16 (deterministic, every call)
  {
    int n1 = L_ * 2 * DI_ * D_;
    f32_to_bf16_kernel<<<(n1 + 255) / 256, 256, 0, stream>>>(in_w, w_in_bf, n1);
    int n2 = L_ * D_ * DI_;
    f32_to_bf16_kernel<<<(n2 + 255) / 256, 256, 0, stream>>>(out_w, w_out_bf, n2);
    int n3 = V_ * D_;
    f32_to_bf16_kernel<<<(n3 + 255) / 256, 256, 0, stream>>>(lm_w, w_lm_bf, n3);
  }

  cond_kernel <<<(B_ * D_ + 255) / 256, 256, 0, stream>>>(latent, cond_w, cond_b, cond);
  embed_kernel<<<(ROWS * D_) / 256,     256, 0, stream>>>(ids, tok_emb, pos_emb, cond, x);

  for (int l = 0; l < L_; ++l) {
    layernorm_bf16_kernel<<<ROWS / 8, 256, 0, stream>>>(x, ln_w + l * D_, ln_b + l * D_, ln_bf);

    dim3 g1(ROWS / 128, XZW / 64);   // 4096x2048, K=512
    gemm_bf16_wmma<<<g1, 256, 0, stream>>>(ln_bf, w_in_bf + (size_t)l * 2 * DI_ * D_,
                                           nullptr, xz, ROWS, XZW, D_);

    conv_silu_kernel<<<(ROWS * DI_) / 256, 256, 0, stream>>>(
        xz, conv_w + (size_t)l * DI_ * DC_, conv_b + l * DI_, xc);

    xproj_kernel<<<(ROWS * XDW + 255) / 256, 256, 0, stream>>>(
        xc, xp_w + (size_t)l * XDW * DI_, xd);

    scan_kernel<<<(B_ * DI_) / 256, 256, 0, stream>>>(
        xc, xd, xz, A_log + (size_t)l * DI_ * DS_, y_bf);

    dim3 g2(ROWS / 128, D_ / 64);    // 4096x512, K=1024, fused residual (in-place x)
    gemm_bf16_wmma<<<g2, 256, 0, stream>>>(y_bf, w_out_bf + (size_t)l * D_ * DI_,
                                           x, x, ROWS, D_, DI_);
  }

  layernorm_bf16_kernel<<<ROWS / 8, 256, 0, stream>>>(x, fln_w, fln_b, ln_bf);

  dim3 g3(ROWS / 128, V_ / 64);      // 4096x128, K=512 -> logits (fp32 output)
  gemm_bf16_wmma<<<g3, 256, 0, stream>>>(ln_bf, w_lm_bf, nullptr, out, ROWS, V_, D_);
}
